// MultiScaleGATEncoder_59931973649017
// MI455X (gfx1250) — matrix operand order repaired and checked by reading, not compiled, not run on hardware
//
#include <hip/hip_runtime.h>
#include <hip/hip_bf16.h>
#include <cstdint>

typedef __attribute__((ext_vector_type(16))) __bf16 bf16x16;
typedef __attribute__((ext_vector_type(8)))  __bf16 bf16x8;
typedef __attribute__((ext_vector_type(8)))  float  f32x8;
typedef __attribute__((ext_vector_type(4)))  unsigned int u32x4;
typedef __attribute__((ext_vector_type(8)))  int i32x8;
typedef __attribute__((ext_vector_type(4)))  int i32x4;

#define LN_EPS 1e-5f

enum {
  EP_RELU_LN = 0,      // Yf,Ybf = LN(relu(XW+b))
  EP_RELU_RES_LN = 1,  // Yf,Ybf = LN(relu(XW+b+res))
  EP_ELU_RES_LN = 2,   // Yf     = LN(res+elu(XW+b))
  EP_BIAS_BF = 3,      // Ybf    = XW+b            (row-major)
  EP_BIAS_BF_T = 4,    // Ybf^T  = XW+b            (n-major, stride M)
  EP_BIAS_F32 = 5      // Yf     = XW+b
};

__device__ __forceinline__ f32x8 wmma_bf16f32(bf16x16 a, bf16x16 b, f32x8 c) {
  return __builtin_amdgcn_wmma_f32_16x16x32_bf16(false, a, false, b, (short)0, c,
                                                 false, false);
}

// A-fragment (16x32 bf16): per lane two contiguous 8-element runs:
//   elems 0..7  <- k = half*16 + e       (16B load)
//   elems 8..15 <- k = half*16 + 8 + e   (16B load)
__device__ __forceinline__ bf16x16 make_a(bf16x8 lo, bf16x8 hi) {
  return __builtin_shufflevector(lo, hi, 0, 1, 2, 3, 4, 5, 6, 7, 8, 9, 10, 11,
                                 12, 13, 14, 15);
}

// ---------------------------------------------------------------------------
// Tensor Data Mover: DMA a 2D bf16 tile (rows x width, row stride `stride`
// elements) from global memory into LDS at byte offset lds_off.
// Descriptor per CDNA5 ISA ch.8: group0 = {flags/count, lds_addr,
// global_addr[31:0], global_addr[56:32]|type=2}; group1 packs data_size=2B,
// tensor/tile dims and dim0 stride; groups 2/3 unused (2D tensor).
// ---------------------------------------------------------------------------
__device__ __forceinline__ void tdm_load_tile_2d(const void* gsrc,
                                                 unsigned lds_off, int width,
                                                 int rows, int stride) {
  const unsigned long long ga = (unsigned long long)(uintptr_t)gsrc;
  u32x4 g0;
  g0[0] = 1u;                                    // count=1 valid descriptor
  g0[1] = lds_off;                               // LDS byte address
  g0[2] = (unsigned)(ga & 0xFFFFFFFFu);          // global_addr[31:0]
  g0[3] = (unsigned)((ga >> 32) & 0x01FFFFFFu)   // global_addr[56:32]
          | (2u << 30);                          // type = 2 ("image")
  i32x8 g1;
  g1[0] = 1 << 16;                               // data_size = 1 -> 2 bytes
  g1[1] = (width & 0xFFFF) << 16;                // tensor_dim0[15:0]
  g1[2] = ((unsigned)width >> 16) | ((rows & 0xFFFF) << 16);  // dim0[31:16]|dim1[15:0]
  g1[3] = (width & 0xFFFF) << 16;                // dim1[31:16]=0 | tile_dim0
  g1[4] = rows & 0xFFFF;                         // tile_dim1 (tile_dim2=0)
  g1[5] = stride;                                // tensor_dim0_stride[31:0]
  g1[6] = 0;
  g1[7] = 0;
  i32x4 gz = {0, 0, 0, 0};
#if __clang_major__ >= 23
  i32x8 z8 = {0, 0, 0, 0, 0, 0, 0, 0};
  __builtin_amdgcn_tensor_load_to_lds(g0, g1, gz, gz, z8, 0);
#else
  __builtin_amdgcn_tensor_load_to_lds(g0, g1, gz, gz, 0);
#endif
}

// ---------------------------------------------------------------------------
// bf16 convert helpers (run once; weights also transposed so B is contiguous)
// ---------------------------------------------------------------------------
__global__ void cvt_bf16_kernel(const float* __restrict__ src,
                                __bf16* __restrict__ dst, int n) {
  const int i = blockIdx.x * 256 + threadIdx.x;
  if (i < n) dst[i] = (__bf16)src[i];
}

__global__ void cvtT_bf16_kernel(const float* __restrict__ W,
                                 __bf16* __restrict__ Wt, int K, int N) {
  const int i = blockIdx.x * 256 + threadIdx.x;
  if (i < K * N) {
    const int k = i / N, n = i % N;
    Wt[(size_t)n * K + k] = (__bf16)W[i];
  }
}

// ---------------------------------------------------------------------------
// Fused GEMM: epilogue(X @ W + b [, res]).  X: MxK bf16 row-major,
// Wt: NxK bf16 (pre-transposed -> B-fragments contiguous).
// The 16-row A tile is TDM-staged into LDS once per block (removes the 8x
// per-wave redundant global reads); A-fragments are ds_load_b128.
// Block = 256 threads (8 waves), tile 16 rows x 512 cols; grid (M/16, N/512).
// LN epilogues require N == 512.
// ---------------------------------------------------------------------------
__global__ void gemm16_kernel(const __bf16* __restrict__ X,
                              const __bf16* __restrict__ Wt,
                              const float* __restrict__ bias,
                              const float* __restrict__ res,
                              const float* __restrict__ gamma,
                              const float* __restrict__ beta,
                              float* __restrict__ Yf,
                              __bf16* __restrict__ Ybf,
                              int M, int Kd, int N, int mode) {
  __shared__ __align__(16) __bf16 Xs[16 * 512];

  const int lane = threadIdx.x & 31;
  const int wave = threadIdx.x >> 5;
  const int half = lane >> 4;
  const int lcol = lane & 15;
  const int row0 = blockIdx.x * 16;
  const int col0 = blockIdx.y * 512 + wave * 64;

  if (wave == 0) {
    tdm_load_tile_2d(X + (size_t)row0 * Kd, (unsigned)(uintptr_t)Xs, Kd, 16, Kd);
    __builtin_amdgcn_s_wait_tensorcnt(0);
  }
  __syncthreads();

  f32x8 acc[4];
#pragma unroll
  for (int t = 0; t < 4; ++t)
#pragma unroll
    for (int r = 0; r < 8; ++r) acc[t][r] = 0.0f;

  const __bf16* Xrow = Xs + lcol * Kd;  // LDS
  for (int kc = 0; kc < Kd; kc += 32) {
    const bf16x8 alo = *(const bf16x8*)(Xrow + kc + half * 16);
    const bf16x8 ahi = *(const bf16x8*)(Xrow + kc + 8 + half * 16);
    const bf16x16 af = make_a(alo, ahi);
    __builtin_prefetch(Wt + (size_t)(col0 + lcol) * Kd + kc + 64, 0, 1);
#pragma unroll
    for (int t = 0; t < 4; ++t) {
      const int n = col0 + t * 16 + lcol;
      const bf16x16 bfr = *(const bf16x16*)(Wt + (size_t)n * Kd + kc + half * 16);
      acc[t] = wmma_bf16f32(af, bfr, acc[t]);
    }
  }

  if (mode >= EP_BIAS_BF) {
#pragma unroll
    for (int t = 0; t < 4; ++t) {
      const int n = col0 + t * 16 + lcol;
#pragma unroll
      for (int r = 0; r < 8; ++r) {
        const int m = row0 + r + half * 8;
        const float v = acc[t][r] + bias[n];
        if (mode == EP_BIAS_BF) Ybf[(size_t)m * N + n] = (__bf16)v;
        else if (mode == EP_BIAS_BF_T) Ybf[(size_t)n * M + m] = (__bf16)v;
        else Yf[(size_t)m * N + n] = v;
      }
    }
    return;
  }

  // ---- LayerNorm epilogues (N == 512) ----
  __shared__ float rsum[16], rsq[16];
  if (threadIdx.x < 16) { rsum[threadIdx.x] = 0.f; rsq[threadIdx.x] = 0.f; }
  __syncthreads();

  float yv[4][8];
  float ps[8], pq[8];
#pragma unroll
  for (int r = 0; r < 8; ++r) { ps[r] = 0.f; pq[r] = 0.f; }
#pragma unroll
  for (int t = 0; t < 4; ++t) {
    const int n = col0 + t * 16 + lcol;
#pragma unroll
    for (int r = 0; r < 8; ++r) {
      const int m = row0 + r + half * 8;
      float v = acc[t][r] + bias[n];
      if (mode == EP_RELU_LN) {
        v = fmaxf(v, 0.f);
      } else if (mode == EP_RELU_RES_LN) {  // relu(xW + b + res)
        v = fmaxf(v + res[m * 512 + n], 0.f);
      } else {                              // res + elu(xW + b)
        const float e = (v > 0.f) ? v : (__expf(v) - 1.f);
        v = res[m * 512 + n] + e;
      }
      yv[t][r] = v;
      ps[r] += v;
      pq[r] += v * v;
    }
  }
#pragma unroll
  for (int r = 0; r < 8; ++r) {
    const int mr = r + half * 8;
    atomicAdd(&rsum[mr], ps[r]);
    atomicAdd(&rsq[mr], pq[r]);
  }
  __syncthreads();

#pragma unroll
  for (int t = 0; t < 4; ++t) {
    const int n = col0 + t * 16 + lcol;
#pragma unroll
    for (int r = 0; r < 8; ++r) {
      const int mr = r + half * 8;
      const int m = row0 + mr;
      const float mean = rsum[mr] * (1.0f / 512.0f);
      const float var = rsq[mr] * (1.0f / 512.0f) - mean * mean;
      const float y =
          (yv[t][r] - mean) * rsqrtf(var + LN_EPS) * gamma[n] + beta[n];
      Yf[m * 512 + n] = y;
      if (Ybf) Ybf[m * 512 + n] = (__bf16)y;
    }
  }
}

// ---------------------------------------------------------------------------
// Flash attention over the complete graph (exact softmax via 2 x 2048-key
// chunks with running max/sum rescale). One block owns 16 query rows and
// loops over the 4 heads; fuses head-mean, +skip, residual and pre-LN.
// LDS: 16x2048 fp32 scores (128KB) + 16x2048 bf16 probs (64KB) +
//      16x2048 bf16 Q tile (64KB, TDM-staged) + scratch  ~= 258KB / 320KB.
// Q,K bf16 row-major; V bf16 TRANSPOSED (d-major) so all WMMA operands are
// contiguous b128 loads.
// ---------------------------------------------------------------------------
#define NN 4096
#define NC 2048
#define HDIM 2048
#define SM_S_BYTES   (16 * NC * 4)
#define SM_P_BYTES   (16 * NC * 2)
#define SM_TMP_BYTES (256 * 4)
#define SM_ST_BYTES  (48 * 4)
#define SM_QS_OFF    (SM_S_BYTES + SM_P_BYTES + SM_TMP_BYTES + SM_ST_BYTES)
#define SM_TOTAL     (SM_QS_OFF + 16 * HDIM * 2)

__global__ void attn_kernel(const __bf16* __restrict__ Qbf,
                            const __bf16* __restrict__ Kbf,
                            const __bf16* __restrict__ Vt,
                            const float* __restrict__ skip,
                            const float* __restrict__ hin,
                            const float* __restrict__ pre_g,
                            const float* __restrict__ pre_b,
                            float* __restrict__ Yf,
                            __bf16* __restrict__ Ybf) {
  extern __shared__ char smraw[];
  float*  S    = (float*)smraw;                            // 16 x NC fp32
  __bf16* Pbf  = (__bf16*)(smraw + SM_S_BYTES);            // 16 x NC bf16
  float*  tmp  = (float*)(smraw + SM_S_BYTES + SM_P_BYTES);  // 256
  float*  mrow = tmp + 256;                                // 16
  float*  lrow = mrow + 16;                                // 16
  float*  arow = lrow + 16;                                // 16 (chunk alphas)
  __bf16* Qs   = (__bf16*)(smraw + SM_QS_OFF);             // 16 x HDIM bf16

  const int lane = threadIdx.x & 31;
  const int wave = threadIdx.x >> 5;
  const int half = lane >> 4;
  const int lcol = lane & 15;
  const int row0 = blockIdx.x * 16;
  const int srow = threadIdx.x >> 4;  // softmax: 16 threads per row
  const int slot = threadIdx.x & 15;
  const float scale = 0.04419417382f;  // 1/sqrt(512)

  // TDM-stage the whole 16 x 2048 Q tile (all heads) into LDS once.
  if (wave == 0) {
    tdm_load_tile_2d(Qbf + (size_t)row0 * HDIM, (unsigned)(uintptr_t)Qs, HDIM,
                     16, HDIM);
    __builtin_amdgcn_s_wait_tensorcnt(0);
  }
  __syncthreads();

  f32x8 osum[4];  // head-mean accumulator (x4, divided at the end)
#pragma unroll
  for (int t = 0; t < 4; ++t)
#pragma unroll
    for (int r = 0; r < 8; ++r) osum[t][r] = 0.0f;

  for (int head = 0; head < 4; ++head) {
    const int hoff = head * 512;

    // Per-head Q fragments from LDS: 16 K-chunks, ds_load_b128 pairs
    bf16x16 qf[16];
    const __bf16* Qrow = Qs + lcol * HDIM + hoff;
#pragma unroll
    for (int kc = 0; kc < 16; ++kc) {
      const bf16x8 lo = *(const bf16x8*)(Qrow + kc * 32 + half * 16);
      const bf16x8 hi = *(const bf16x8*)(Qrow + kc * 32 + 8 + half * 16);
      qf[kc] = make_a(lo, hi);
    }

    if (threadIdx.x < 16) { mrow[threadIdx.x] = -3.0e38f; lrow[threadIdx.x] = 0.f; }
    f32x8 oh[4];
#pragma unroll
    for (int t = 0; t < 4; ++t)
#pragma unroll
      for (int r = 0; r < 8; ++r) oh[t][r] = 0.0f;
    __syncthreads();

    for (int chunk = 0; chunk < 2; ++chunk) {
      const int kb = chunk * NC;

      // Phase 1: S = (Q K^T)*scale for this chunk; each wave does 256 keys
      for (int kt = 0; kt < 16; ++kt) {
        const int jloc = wave * 256 + kt * 16;  // chunk-local key of lcol tile
        const int key = kb + jloc + lcol;
        f32x8 sc;
#pragma unroll
        for (int r = 0; r < 8; ++r) sc[r] = 0.0f;
        const __bf16* Krow = Kbf + (size_t)key * HDIM + hoff;
#pragma unroll
        for (int kc = 0; kc < 16; ++kc) {
          const bf16x16 bfr = *(const bf16x16*)(Krow + kc * 32 + half * 16);
          sc = wmma_bf16f32(qf[kc], bfr, sc);
        }
#pragma unroll
        for (int r = 0; r < 8; ++r)
          S[(r + half * 8) * NC + jloc + lcol] = sc[r] * scale;
      }
      __syncthreads();

      // Phase 2: chunk max/sum + running rescale; write probs as bf16
      float mx = -3.0e38f;
      for (int j = slot; j < NC; j += 16) mx = fmaxf(mx, S[srow * NC + j]);
      tmp[threadIdx.x] = mx;
      __syncthreads();
      for (int off = 8; off > 0; off >>= 1) {
        if (slot < off)
          tmp[threadIdx.x] = fmaxf(tmp[threadIdx.x], tmp[threadIdx.x + off]);
        __syncthreads();
      }
      const float m_old = mrow[srow];
      const float m_new = fmaxf(m_old, tmp[srow * 16]);
      __syncthreads();
      float ssum = 0.f;
      for (int j = slot; j < NC; j += 16) {
        const float e = __expf(S[srow * NC + j] - m_new);
        Pbf[srow * NC + j] = (__bf16)e;
        ssum += e;
      }
      tmp[threadIdx.x] = ssum;
      __syncthreads();
      for (int off = 8; off > 0; off >>= 1) {
        if (slot < off) tmp[threadIdx.x] += tmp[threadIdx.x + off];
        __syncthreads();
      }
      if (slot == 0) {
        const float alpha = __expf(m_old - m_new);
        lrow[srow] = lrow[srow] * alpha + tmp[srow * 16];
        mrow[srow] = m_new;
        arow[srow] = alpha;
      }
      __syncthreads();

      // Phase 3: O = O*alpha + P @ V_chunk; wave owns 64 output cols
      float al[8];
#pragma unroll
      for (int r = 0; r < 8; ++r) al[r] = arow[r + half * 8];
#pragma unroll
      for (int t = 0; t < 4; ++t)
#pragma unroll
        for (int r = 0; r < 8; ++r) oh[t][r] *= al[r];

      const __bf16* Prow = Pbf + lcol * NC;
      for (int kc = 0; kc < NC; kc += 32) {
        const bf16x8 plo = *(const bf16x8*)(Prow + kc + half * 16);
        const bf16x8 phi = *(const bf16x8*)(Prow + kc + 8 + half * 16);
        const bf16x16 pf = make_a(plo, phi);
#pragma unroll
        for (int t = 0; t < 4; ++t) {
          const int n = wave * 64 + t * 16 + lcol;
          const bf16x16 bfr =
              *(const bf16x16*)(Vt + (size_t)(hoff + n) * NN + kb + kc + half * 16);
          oh[t] = wmma_bf16f32(pf, bfr, oh[t]);
        }
      }
      __syncthreads();  // protect S/Pbf before next chunk
    }

    // Normalize by running sum, accumulate head
    float li[8];
#pragma unroll
    for (int r = 0; r < 8; ++r) li[r] = 1.0f / lrow[r + half * 8];
#pragma unroll
    for (int t = 0; t < 4; ++t)
#pragma unroll
      for (int r = 0; r < 8; ++r) osum[t][r] += oh[t][r] * li[r];
    __syncthreads();  // lrow reads done before next head resets stats
  }

  // Epilogue: attn = mean_heads + skip; y = LN(h + attn, pre_g, pre_b)
  if (threadIdx.x < 32) tmp[threadIdx.x] = 0.f;
  __syncthreads();
  float yv[4][8];
  float ps[8], pq[8];
#pragma unroll
  for (int r = 0; r < 8; ++r) { ps[r] = 0.f; pq[r] = 0.f; }
#pragma unroll
  for (int t = 0; t < 4; ++t) {
    const int n = wave * 64 + t * 16 + lcol;
#pragma unroll
    for (int r = 0; r < 8; ++r) {
      const int m = row0 + r + half * 8;
      const float a = osum[t][r] * 0.25f + skip[m * 512 + n];
      const float v = hin[m * 512 + n] + a;
      yv[t][r] = v;
      ps[r] += v;
      pq[r] += v * v;
    }
  }
#pragma unroll
  for (int r = 0; r < 8; ++r) {
    const int mr = r + half * 8;
    atomicAdd(&tmp[mr], ps[r]);
    atomicAdd(&tmp[16 + mr], pq[r]);
  }
  __syncthreads();
#pragma unroll
  for (int t = 0; t < 4; ++t) {
    const int n = wave * 64 + t * 16 + lcol;
#pragma unroll
    for (int r = 0; r < 8; ++r) {
      const int mr = r + half * 8;
      const int m = row0 + mr;
      const float mean = tmp[mr] * (1.0f / 512.0f);
      const float var = tmp[16 + mr] * (1.0f / 512.0f) - mean * mean;
      const float y =
          (yv[t][r] - mean) * rsqrtf(var + LN_EPS) * pre_g[n] + pre_b[n];
      Yf[m * 512 + n] = y;
      Ybf[m * 512 + n] = (__bf16)y;
    }
  }
}

// ---------------------------------------------------------------------------
extern "C" void kernel_launch(void* const* d_in, const int* in_sizes, int n_in,
                              void* d_out, int out_size, void* d_ws, size_t ws_size,
                              hipStream_t stream) {
  (void)in_sizes; (void)n_in; (void)out_size; (void)ws_size;
  const float* x      = (const float*)d_in[0];
  const float* W1     = (const float*)d_in[1];
  const float* b1     = (const float*)d_in[2];
  const float* ln1_g  = (const float*)d_in[3];
  const float* ln1_b  = (const float*)d_in[4];
  const float* W2     = (const float*)d_in[5];
  const float* b2     = (const float*)d_in[6];
  const float* ln2_g  = (const float*)d_in[7];
  const float* ln2_b  = (const float*)d_in[8];
  const float* Wq     = (const float*)d_in[9];
  const float* bq     = (const float*)d_in[10];
  const float* Wk     = (const float*)d_in[11];
  const float* bk     = (const float*)d_in[12];
  const float* Wv     = (const float*)d_in[13];
  const float* bv     = (const float*)d_in[14];
  const float* Wsk    = (const float*)d_in[15];
  const float* bsk    = (const float*)d_in[16];
  const float* pre_g  = (const float*)d_in[17];
  const float* pre_b  = (const float*)d_in[18];
  const float* Wf     = (const float*)d_in[19];
  const float* bff    = (const float*)d_in[20];
  const float* post_g = (const float*)d_in[21];
  const float* post_b = (const float*)d_in[22];

  const int N = 4096, D = 512, INDIM = 256, HD = 2048;
  char* ws = (char*)d_ws;
  size_t off = 0;
  auto carve = [&](size_t bytes) {
    void* p = ws + off;
    off += (bytes + 255) & ~(size_t)255;
    return p;
  };
  // activations
  __bf16* xbf   = (__bf16*)carve((size_t)N * INDIM * 2);   // 2 MB
  float*  h1f   = (float*)carve((size_t)N * D * 4);        // 8 MB
  __bf16* h1bf  = (__bf16*)carve((size_t)N * D * 2);       // 4 MB
  float*  hBf   = (float*)carve((size_t)N * D * 4);        // 8 MB
  __bf16* hBbf  = (__bf16*)carve((size_t)N * D * 2);       // 4 MB
  __bf16* qbf   = (__bf16*)carve((size_t)N * HD * 2);      // 16 MB
  __bf16* kbf   = (__bf16*)carve((size_t)N * HD * 2);      // 16 MB
  __bf16* vtbf  = (__bf16*)carve((size_t)HD * N * 2);      // 16 MB (transposed)
  float*  skf   = (float*)carve((size_t)N * D * 4);        // 8 MB
  float*  h2f   = (float*)carve((size_t)N * D * 4);        // 8 MB
  __bf16* h2bf  = (__bf16*)carve((size_t)N * D * 2);       // 4 MB
  // transposed bf16 weights
  __bf16* W1t = (__bf16*)carve((size_t)INDIM * D * 2);
  __bf16* W2t = (__bf16*)carve((size_t)D * D * 2);
  __bf16* Wqt = (__bf16*)carve((size_t)D * HD * 2);
  __bf16* Wkt = (__bf16*)carve((size_t)D * HD * 2);
  __bf16* Wvt = (__bf16*)carve((size_t)D * HD * 2);
  __bf16* Wst = (__bf16*)carve((size_t)D * D * 2);
  __bf16* Wft = (__bf16*)carve((size_t)D * D * 2);
  float* out = (float*)d_out;

  dim3 blk(256);
  // one-time bf16 conversion / weight transposition
  cvt_bf16_kernel<<<dim3((N * INDIM + 255) / 256), blk, 0, stream>>>(x, xbf, N * INDIM);
  cvtT_bf16_kernel<<<dim3((INDIM * D + 255) / 256), blk, 0, stream>>>(W1, W1t, INDIM, D);
  cvtT_bf16_kernel<<<dim3((D * D + 255) / 256), blk, 0, stream>>>(W2, W2t, D, D);
  cvtT_bf16_kernel<<<dim3((D * HD + 255) / 256), blk, 0, stream>>>(Wq, Wqt, D, HD);
  cvtT_bf16_kernel<<<dim3((D * HD + 255) / 256), blk, 0, stream>>>(Wk, Wkt, D, HD);
  cvtT_bf16_kernel<<<dim3((D * HD + 255) / 256), blk, 0, stream>>>(Wv, Wvt, D, HD);
  cvtT_bf16_kernel<<<dim3((D * D + 255) / 256), blk, 0, stream>>>(Wsk, Wst, D, D);
  cvtT_bf16_kernel<<<dim3((D * D + 255) / 256), blk, 0, stream>>>(Wf, Wft, D, D);

  // mlp1
  gemm16_kernel<<<dim3(N / 16, 1), blk, 0, stream>>>(
      xbf, W1t, b1, nullptr, ln1_g, ln1_b, h1f, h1bf, N, INDIM, D, EP_RELU_LN);
  gemm16_kernel<<<dim3(N / 16, 1), blk, 0, stream>>>(
      h1bf, W2t, b2, h1f, ln2_g, ln2_b, hBf, hBbf, N, D, D, EP_RELU_RES_LN);
  // q/k/v/skip projections (q,k row-major bf16; v transposed bf16; skip f32)
  gemm16_kernel<<<dim3(N / 16, 4), blk, 0, stream>>>(
      hBbf, Wqt, bq, nullptr, nullptr, nullptr, nullptr, qbf, N, D, HD, EP_BIAS_BF);
  gemm16_kernel<<<dim3(N / 16, 4), blk, 0, stream>>>(
      hBbf, Wkt, bk, nullptr, nullptr, nullptr, nullptr, kbf, N, D, HD, EP_BIAS_BF);
  gemm16_kernel<<<dim3(N / 16, 4), blk, 0, stream>>>(
      hBbf, Wvt, bv, nullptr, nullptr, nullptr, nullptr, vtbf, N, D, HD, EP_BIAS_BF_T);
  gemm16_kernel<<<dim3(N / 16, 1), blk, 0, stream>>>(
      hBbf, Wst, bsk, nullptr, nullptr, nullptr, skf, nullptr, N, D, D, EP_BIAS_F32);

  // flash attention + head-mean + skip + residual + pre-LN
  hipFuncSetAttribute((const void*)attn_kernel,
                      hipFuncAttributeMaxDynamicSharedMemorySize, (int)SM_TOTAL);
  attn_kernel<<<dim3(N / 16), blk, (size_t)SM_TOTAL, stream>>>(
      qbf, kbf, vtbf, skf, hBf, pre_g, pre_b, h2f, h2bf);

  // feed-forward (ELU) + residual + post-LN -> final output
  gemm16_kernel<<<dim3(N / 16, 1), blk, 0, stream>>>(
      h2bf, Wft, bff, h2f, post_g, post_b, out, nullptr, N, D, D, EP_ELU_RES_LN);
}
